// NetVLAD_31516470018435
// MI455X (gfx1250) — matrix operand order repaired and checked by reading, not compiled, not run on hardware
//
#include <hip/hip_runtime.h>

typedef __attribute__((ext_vector_type(8)))  _Float16 v8h;
typedef __attribute__((ext_vector_type(16))) _Float16 v16h;
typedef __attribute__((ext_vector_type(8)))  float    v8f;

#define NIMG 32
#define DCH  512
#define SPX  1600
#define KCL  64

__device__ __forceinline__ v16h make_v16(v8h a, v8h b){
    return __builtin_shufflevector(a, b, 0,1,2,3,4,5,6,7,8,9,10,11,12,13,14,15);
}

// A-matrix 16x32 f16 fragment (row-major memory, leading dim `ld` in elements).
// Lane L<16 holds row M=L, K={0..15}; lane L>=16 holds row M=L-16, K={16..31}.
__device__ __forceinline__ v16h load_a16(const _Float16* __restrict__ base, int ld){
    int lane = threadIdx.x & 31;
    const _Float16* p = base + (size_t)(lane & 15) * ld + ((lane >> 4) << 4);
    v8h x0 = *(const v8h*)p;
    v8h x1 = *(const v8h*)(p + 8);
    return make_v16(x0, x1);
}

// B-matrix 32x16 f16 fragment. Memory: column n starts at base + n*ld, K contiguous.
// Lane L holds column N=L&15, K = (L>=16 ? 16 : 0) .. +15 contiguous.
__device__ __forceinline__ v16h load_b16(const _Float16* __restrict__ base, int ld){
    int lane = threadIdx.x & 31;
    const _Float16* p = base + (size_t)(lane & 15) * ld + ((lane >> 4) << 4);
    v8h x0 = *(const v8h*)p;
    v8h x1 = *(const v8h*)(p + 8);
    return make_v16(x0, x1);
}

// ---------------- K0: convert conv_weight to f16 ----------------
__global__ void k0_cvt(const float* __restrict__ w, _Float16* __restrict__ wh){
    int i = blockIdx.x * 256 + threadIdx.x;   // 64*512 = 32768 elements
    wh[i] = (_Float16)w[i];
}

// ---------------- K1: per-pixel L2 norm + dual-layout f16 write ----------------
__global__ void k1_norm(const float* __restrict__ x,
                        _Float16* __restrict__ xsd,   // [n][s][d]  (d contiguous)
                        _Float16* __restrict__ xds){  // [n][d][s]  (s contiguous)
    __shared__ float    part[4][64];
    __shared__ float    inv[64];
    __shared__ _Float16 tp[64][66];   // 33*d mod 64 is a permutation -> conflict-free transpose reads
    int t  = threadIdx.x;
    int sx = t & 63, ty = t >> 6;
    int n  = blockIdx.x / 25;
    int s0 = (blockIdx.x % 25) * 64;

    const float* xb = x + (size_t)n * DCH * SPX + s0 + sx;

    float acc = 0.f;
    for (int d = ty; d < DCH; d += 4){ float v = xb[(size_t)d * SPX]; acc += v * v; }
    part[ty][sx] = acc;
    __syncthreads();
    if (t < 64){
        float s = part[0][t] + part[1][t] + part[2][t] + part[3][t];
        inv[t] = 1.f / fmaxf(sqrtf(s), 1e-12f);
    }
    __syncthreads();
    float iv = inv[sx];

    int dcol = t & 63, sr = t >> 6;
    for (int db = 0; db < DCH; db += 64){
        #pragma unroll
        for (int j = 0; j < 16; ++j){
            int dl = j * 4 + ty;
            int d  = db + dl;
            _Float16 h = (_Float16)(xb[(size_t)d * SPX] * iv);
            xds[(size_t)(n * DCH + d) * SPX + s0 + sx] = h;   // coalesced along s
            tp[dl][sx] = h;
        }
        __syncthreads();
        #pragma unroll
        for (int r = 0; r < 16; ++r){
            int srow = sr + r * 4;
            xsd[(size_t)(n * SPX + s0 + srow) * DCH + db + dcol] = tp[dcol][srow]; // coalesced along d
        }
        __syncthreads();
    }
}

__device__ __forceinline__ void async_issue(unsigned loff, const _Float16* g){
    asm volatile("global_load_async_to_lds_b128 %0, %1, off" :: "v"(loff), "v"(g) : "memory");
}

// Async-copy one 32-pixel x 128-d f16 chunk (8 KB) global -> LDS.
// Branchless: each of the 256 threads issues exactly two b128 async loads (ASYNCcnt +2 per wave).
__device__ __forceinline__ void async_copy_chunk(_Float16 (*dst)[136],
                                                 const _Float16* __restrict__ src,
                                                 int dbase, int t){
    int s0 = t >> 4;              // pixel row 0..15 (second issue covers +16)
    int dd = (t & 15) << 3;       // 16-byte unit within the 128-d chunk
    const _Float16* g = src + (size_t)s0 * DCH + dbase + dd;   // coalesced 16B/lane
    async_issue((unsigned)(size_t)&dst[s0][dd],      g);
    async_issue((unsigned)(size_t)&dst[s0 + 16][dd], g + (size_t)16 * DCH);
}

// ---------------- K2: logits (WMMA, LDS-staged B via async copy) + softmax ----------------
// One block = (image n, 32-pixel tile). 8 waves: wave w -> k-tile w&3, s-subtile w>>2.
__global__ void k2_logits(const _Float16* __restrict__ wh,
                          const _Float16* __restrict__ xsd,
                          _Float16* __restrict__ asg){
    __shared__ __align__(16) _Float16 bt[2][32][136]; // double-buffered B tile; 68 dw ≡ 4 mod 64 banks
    __shared__ float lg[KCL][33];
    int t    = threadIdx.x;
    int n    = blockIdx.x / 50;
    int s0   = (blockIdx.x % 50) * 32;
    int wave = t >> 5, lane = t & 31;
    int kt   = wave & 3, st = wave >> 2;

    const _Float16* xrow = xsd + (size_t)(n * SPX + s0) * DCH;
    const _Float16* arow = wh  + (size_t)(kt * 16) * DCH;

    async_copy_chunk(bt[0], xrow, 0, t);

    v8f c = {};
    for (int ch = 0; ch < 4; ++ch){
        if (ch < 3){
            async_copy_chunk(bt[(ch + 1) & 1], xrow, (ch + 1) * 128, t);
            // async loads complete in order: <=2 outstanding  => chunk ch landed
            asm volatile("s_wait_asynccnt 0x2" ::: "memory");
        } else {
            asm volatile("s_wait_asynccnt 0x0" ::: "memory");
        }
        __syncthreads();
        const _Float16* brow = &bt[ch & 1][st * 16 + (lane & 15)][(lane >> 4) << 4];
        #pragma unroll
        for (int j = 0; j < 4; ++j){
            v16h a = load_a16(arow + ch * 128 + j * 32, DCH);
            const _Float16* p = brow + j * 32;                 // ds_load_b128 x2, conflict-free
            v16h b = make_v16(*(const v8h*)p, *(const v8h*)(p + 8));
            c = __builtin_amdgcn_wmma_f32_16x16x32_f16(false, a, false, b, (short)0, c, false, false);
        }
        __syncthreads();
    }

    int col = lane & 15, r0 = (lane >> 4) * 8;
    #pragma unroll
    for (int i = 0; i < 8; ++i) lg[kt * 16 + r0 + i][st * 16 + col] = c[i];
    __syncthreads();

    if (t < 32){                       // one wave: softmax over K=64 for its 32 pixels
        int p = t;
        float mx = -3.4e38f;
        for (int k = 0; k < KCL; ++k) mx = fmaxf(mx, lg[k][p]);
        float sum = 0.f;
        for (int k = 0; k < KCL; ++k){ float e = __expf(lg[k][p] - mx); lg[k][p] = e; sum += e; }
        float isum = 1.f / sum;
        for (int k = 0; k < KCL; ++k)  // lanes p contiguous -> coalesced per k-row
            asg[(size_t)(n * KCL + k) * SPX + s0 + p] = (_Float16)(lg[k][p] * isum);
    }
}

// ---------------- K3: agg = a @ x̂ᵀ (WMMA over full s) ----------------
__global__ void k3_agg(const _Float16* __restrict__ asg,
                       const _Float16* __restrict__ xds,
                       float* __restrict__ agg){
    int t    = threadIdx.x;
    int n    = blockIdx.x / 16;
    int db   = blockIdx.x % 16;
    int wave = t >> 5, lane = t & 31;
    int kt   = wave & 3;
    int dt   = db * 2 + (wave >> 2);

    const _Float16* arow = asg + (size_t)(n * KCL + kt * 16) * SPX;
    const _Float16* brow = xds + (size_t)(n * DCH + dt * 16) * SPX;

    v8f c = {};
    #pragma unroll 2
    for (int s0 = 0; s0 < SPX; s0 += 32){
        v16h a = load_a16(arow + s0, SPX);
        v16h b = load_b16(brow + s0, SPX);
        c = __builtin_amdgcn_wmma_f32_16x16x32_f16(false, a, false, b, (short)0, c, false, false);
    }
    int col = lane & 15, r0 = (lane >> 4) * 8;
    #pragma unroll
    for (int i = 0; i < 8; ++i)
        agg[(size_t)(n * KCL + kt * 16 + r0 + i) * DCH + dt * 16 + col] = c[i];
}

// ---------------- K3b: a_sum[n,k] = sum_s a ----------------
__global__ void k3b_asum(const _Float16* __restrict__ asg, float* __restrict__ a_sum){
    __shared__ float red[256];
    int nk = blockIdx.x, t = threadIdx.x;
    const _Float16* p = asg + (size_t)nk * SPX;
    float acc = 0.f;
    for (int s = t; s < SPX; s += 256) acc += (float)p[s];
    red[t] = acc; __syncthreads();
    for (int w = 128; w > 0; w >>= 1){ if (t < w) red[t] += red[t + w]; __syncthreads(); }
    if (t == 0) a_sum[nk] = red[0];
}

// ---------------- K4a: vlad = agg - a_sum*c, intra-normalize over d ----------------
__global__ void k4a_intra(const float* __restrict__ agg, const float* __restrict__ a_sum,
                          const float* __restrict__ cent, float* __restrict__ out){
    __shared__ float red[256];
    int nk = blockIdx.x, t = threadIdx.x;
    int n = nk >> 6, k = nk & 63;
    float as = a_sum[nk];
    float v0 = agg[(size_t)nk * DCH + t]       - as * cent[k * DCH + t];
    float v1 = agg[(size_t)nk * DCH + t + 256] - as * cent[k * DCH + t + 256];
    red[t] = v0 * v0 + v1 * v1; __syncthreads();
    for (int w = 128; w > 0; w >>= 1){ if (t < w) red[t] += red[t + w]; __syncthreads(); }
    float inv = 1.f / fmaxf(sqrtf(red[0]), 1e-12f);
    float* o = out + (size_t)n * KCL * DCH + (size_t)k * DCH;
    o[t] = v0 * inv; o[t + 256] = v1 * inv;
}

// ---------------- K4b/K4c: global L2 normalization per image ----------------
__global__ void k4b_gsum(const float* __restrict__ out, float* __restrict__ invg){
    __shared__ float red[256];
    int n = blockIdx.x, t = threadIdx.x;
    const float* p = out + (size_t)n * KCL * DCH;
    float acc = 0.f;
    for (int i = t; i < KCL * DCH; i += 256){ float v = p[i]; acc += v * v; }
    red[t] = acc; __syncthreads();
    for (int w = 128; w > 0; w >>= 1){ if (t < w) red[t] += red[t + w]; __syncthreads(); }
    if (t == 0) invg[n] = 1.f / fmaxf(sqrtf(red[0]), 1e-12f);
}
__global__ void k4c_scale(float* __restrict__ out, const float* __restrict__ invg){
    size_t i = (size_t)blockIdx.x * 256 + threadIdx.x;
    out[i] *= invg[i >> 15];   // 32768 elements per image
}

extern "C" void kernel_launch(void* const* d_in, const int* in_sizes, int n_in,
                              void* d_out, int out_size, void* d_ws, size_t ws_size,
                              hipStream_t stream){
    (void)in_sizes; (void)n_in; (void)out_size; (void)ws_size;
    const float* x    = (const float*)d_in[0];   // [32,512,40,40]
    const float* w    = (const float*)d_in[1];   // [64,512]
    const float* cent = (const float*)d_in[2];   // [64,512]
    float* out = (float*)d_out;                  // [32, 32768]

    char* ws = (char*)d_ws;
    const size_t SZ_WH  = 65536;                         // 64*512 f16
    const size_t SZ_X   = (size_t)NIMG * DCH * SPX * 2;  // 52,428,800
    const size_t SZ_ASG = (size_t)NIMG * KCL * SPX * 2;  // 6,553,600
    _Float16* wh    = (_Float16*)(ws);
    _Float16* xsd   = (_Float16*)(ws + SZ_WH);
    _Float16* xds   = (_Float16*)(ws + SZ_WH + SZ_X);
    _Float16* asg   = (_Float16*)(ws + SZ_WH + 2 * SZ_X);
    float*    a_sum = (float*)   (ws + SZ_WH + 2 * SZ_X + SZ_ASG);
    float*    agg   = (float*)   (ws + SZ_WH + 2 * SZ_X + SZ_ASG + 8192);
    float*    invg  = (float*)   (ws + SZ_WH + 2 * SZ_X + SZ_ASG + 8192 + (size_t)NIMG * KCL * DCH * 4);

    k0_cvt   <<<128,            256, 0, stream>>>(w, wh);
    k1_norm  <<<NIMG * 25,      256, 0, stream>>>(x, xsd, xds);
    k2_logits<<<NIMG * 50,      256, 0, stream>>>(wh, xsd, asg);
    k3_agg   <<<NIMG * 16,      256, 0, stream>>>(asg, xds, agg);
    k3b_asum <<<NIMG * KCL,     256, 0, stream>>>(asg, a_sum);
    k4a_intra<<<NIMG * KCL,     256, 0, stream>>>(agg, a_sum, cent, out);
    k4b_gsum <<<NIMG,           256, 0, stream>>>(out, invg);
    k4c_scale<<<NIMG * KCL * DCH / 256, 256, 0, stream>>>(out, invg);
}
